// NeuralMemory_46437186404632
// MI455X (gfx1250) — matrix operand order repaired and checked by reading, not compiled, not run on hardware
//
#include <hip/hip_runtime.h>
#include <hip/hip_bf16.h>

// Problem dims (fixed by reference): B=64, S=8192, E=32, H=64
#define B_DIM 64
#define S_LEN 8192
#define E_DIM 32
#define H_DIM 64
#define NW 8              // waves per block
#define TILE_S (16 * NW)  // 128 s-rows per block
#define H_STRIDE 72       // padded LDS row stride (elements) for h tile

typedef __attribute__((ext_vector_type(16))) __bf16 v16bf;
typedef __attribute__((ext_vector_type(8)))  __bf16 v8bf;
typedef __attribute__((ext_vector_type(8)))  float  v8f;
typedef __attribute__((ext_vector_type(4)))  float  f32x4;

__device__ __forceinline__ float  tof(__bf16 h) { return (float)h; }
__device__ __forceinline__ __bf16 tobf(float f) { return (__bf16)f; }
// silu via v_exp_f32 + v_rcp_f32 (avoid the IEEE div_scale/div_fmas chain)
__device__ __forceinline__ float  siluf(float v) {
    return v * __builtin_amdgcn_rcpf(1.0f + __expf(-v));
}

// ---------------------------------------------------------------------------
// Kernel 0: zero the sumsq accumulator (B*E floats)
// ---------------------------------------------------------------------------
__global__ void nm_zero(float* __restrict__ p, int n) {
    int i = blockIdx.x * blockDim.x + threadIdx.x;
    if (i < n) p[i] = 0.0f;
}

// ---------------------------------------------------------------------------
// Pass 1: q = silu(x @ Qw^T)  (bf16 WMMA, K=32), store q bf16 to workspace,
//         accumulate sum_s q^2 per (b, e) via LDS + global f32 atomics.
// grid = (S/TILE_S, B), block = 32*NW
// ---------------------------------------------------------------------------
__global__ void __launch_bounds__(32 * NW)
nm_pass1(const float* __restrict__ x, const float* __restrict__ Qw,
         __hip_bfloat16* __restrict__ qws_, float* __restrict__ sumsq) {
    __shared__ float lsum[E_DIM];
    __bf16* qws = reinterpret_cast<__bf16*>(qws_);

    const int tid  = threadIdx.x;
    const int lane = tid & 31;
    const int wid  = tid >> 5;
    const int b    = blockIdx.y;
    const int s0   = blockIdx.x * TILE_S + wid * 16;
    const int half = lane >> 4;    // K-interleave half
    const int row  = lane & 15;    // M row (A) / N column (B, D)

    if (tid < E_DIM) lsum[tid] = 0.0f;
    __syncthreads();

    // ---- A fragment: x rows (f32 -> bf16), 16x32, ISA 16-bit A layout ----
    // lanes 0-15 : row M=lane,    K = 0..7  and 16..23
    // lanes 16-31: row M=lane-16, K = 8..15 and 24..31
    const float* xrow = x + ((size_t)b * S_LEN + (s0 + row)) * E_DIM + half * 8;
    f32x4 xa = *(const f32x4*)(xrow);
    f32x4 xb = *(const f32x4*)(xrow + 4);
    f32x4 xc = *(const f32x4*)(xrow + 16);
    f32x4 xd = *(const f32x4*)(xrow + 20);
    v16bf a;
#pragma unroll
    for (int j = 0; j < 4; ++j) {
        a[j]      = tobf(xa[j]);
        a[4 + j]  = tobf(xb[j]);
        a[8 + j]  = tobf(xc[j]);
        a[12 + j] = tobf(xd[j]);
    }

    // per-lane output base: rows m = 8*half .. 8*half+7, column f varies by g
    __bf16* qbase = qws + ((size_t)b * S_LEN + s0 + 8 * half) * E_DIM + row;

    // ---- two N-groups of 16 output features each ----
#pragma unroll
    for (int g = 0; g < 2; ++g) {
        // B fragment (dense 32x16 bf16): lane n holds column f = 16g + n,
        // lanes 0-15 hold K=0..7,16..23, lanes 16-31 hold K=8..15,24..31.
        const float* qwrow = Qw + (size_t)(16 * g + row) * E_DIM + 8 * half;
        v16bf bb;
#pragma unroll
        for (int j = 0; j < 16; j += 4) {
            f32x4 t = *(const f32x4*)(qwrow + j + (j & 8));
            bb[j] = tobf(t[0]); bb[j + 1] = tobf(t[1]);
            bb[j + 2] = tobf(t[2]); bb[j + 3] = tobf(t[3]);
        }

        v8f c = {};
        c = __builtin_amdgcn_wmma_f32_16x16x32_bf16(false, a, false, bb,
                                                    (short)0, c, false, false);

        // D layout: lane holds column f = 16g + (lane&15), rows m = r + 8*half
        __bf16* qp = qbase + 16 * g;
        float part = 0.0f;
#pragma unroll
        for (int r = 0; r < 8; ++r) {
            float v = siluf(c[r]);
            part += v * v;
            qp[r * E_DIM] = tobf(v);      // constant 64B-stride offsets
        }
        atomicAdd(&lsum[16 * g + row], part);   // ds_add_f32
    }

    __syncthreads();
    if (tid < E_DIM) atomicAdd(&sumsq[b * E_DIM + tid], lsum[tid]);
}

// ---------------------------------------------------------------------------
// Pass 2: scale q by 1/max(||q||_S, eps), h = silu(q@W1^T + b1),
//         out = h@W2^T + b2.  bf16 WMMA throughout, f32 accumulate.
// grid = (S/TILE_S, B), block = 32*NW
// ---------------------------------------------------------------------------
__global__ void __launch_bounds__(32 * NW)
nm_pass2(const __hip_bfloat16* __restrict__ qws_, const float* __restrict__ sumsq,
         const float* __restrict__ W1, const float* __restrict__ b1,
         const float* __restrict__ W2, const float* __restrict__ b2,
         float* __restrict__ out) {
    __shared__ float  scale[E_DIM];
    __shared__ __bf16 hlds[NW][16 * H_STRIDE];
    const __bf16* qws = reinterpret_cast<const __bf16*>(qws_);

    const int tid  = threadIdx.x;
    const int lane = tid & 31;
    const int wid  = tid >> 5;
    const int b    = blockIdx.y;
    const int s0   = blockIdx.x * TILE_S + wid * 16;
    const int half = lane >> 4;
    const int row  = lane & 15;

    // warm GL2/L0 with this batch's W2 block (emits global_prefetch_b8)
    __builtin_prefetch(W2 + (size_t)b * E_DIM * H_DIM + tid * 8, 0, 1);

    if (tid < E_DIM) {
        float n = __builtin_sqrtf(sumsq[b * E_DIM + tid]);
        scale[tid] = __builtin_amdgcn_rcpf(fmaxf(n, 1e-12f));
    }
    __syncthreads();

    // ---- A fragment: q row (bf16), column-scaled by inv-norm ----
    const __bf16* qrow = qws + ((size_t)b * S_LEN + (s0 + row)) * E_DIM + half * 8;
    v8bf qa = *(const v8bf*)(qrow);
    v8bf qb = *(const v8bf*)(qrow + 16);
    v16bf a;
#pragma unroll
    for (int j = 0; j < 8; ++j) {
        a[j]     = tobf(tof(qa[j]) * scale[half * 8 + j]);
        a[8 + j] = tobf(tof(qb[j]) * scale[16 + half * 8 + j]);
    }

    // ---- GEMM1: h = silu(q @ W1^T + b1), H=64 -> 4 N-groups ----
    __bf16* htile = hlds[wid];
    __bf16* hbase = htile + 8 * half * H_STRIDE + row;   // rows m=8*half.., col h
#pragma unroll
    for (int g = 0; g < 4; ++g) {
        const int h = 16 * g + row;
        const float* w1row = W1 + ((size_t)b * H_DIM + h) * E_DIM + 8 * half;
        v16bf bb;
#pragma unroll
        for (int j = 0; j < 16; j += 4) {
            f32x4 t = *(const f32x4*)(w1row + j + (j & 8));
            bb[j] = tobf(t[0]); bb[j + 1] = tobf(t[1]);
            bb[j + 2] = tobf(t[2]); bb[j + 3] = tobf(t[3]);
        }
        v8f c = {};
        c = __builtin_amdgcn_wmma_f32_16x16x32_bf16(false, a, false, bb,
                                                    (short)0, c, false, false);
        const float bias = b1[b * H_DIM + h];
        // D lane holds column h; scatter rows m into row-major LDS h tile
        __bf16* hp = hbase + 16 * g;
#pragma unroll
        for (int r = 0; r < 8; ++r) {
            hp[r * H_STRIDE] = tobf(siluf(c[r] + bias));  // constant 144B offsets
        }
    }
    __syncthreads();   // D-layout -> A-layout transpose through LDS

    // ---- reload h as A fragments (16x64 = two 16x32 K-chunks) ----
    const __bf16* hrow = htile + row * H_STRIDE + half * 8;
    v8bf h0 = *(const v8bf*)(hrow);
    v8bf h1 = *(const v8bf*)(hrow + 16);
    v8bf h2 = *(const v8bf*)(hrow + 32);
    v8bf h3 = *(const v8bf*)(hrow + 48);
    v16bf a2a, a2b;
#pragma unroll
    for (int j = 0; j < 8; ++j) {
        a2a[j] = h0[j]; a2a[8 + j] = h1[j];
        a2b[j] = h2[j]; a2b[8 + j] = h3[j];
    }

    // per-lane output base pointer: constant offsets per (g2, r)
    float* obase = out + ((size_t)b * S_LEN + s0 + 8 * half) * E_DIM + row;

    // ---- GEMM2: out = h @ W2^T + b2, K=64 -> 2 chained WMMAs, E=32 -> 2 groups
#pragma unroll
    for (int g2 = 0; g2 < 2; ++g2) {
        const int e = 16 * g2 + row;
        const float* w2row = W2 + ((size_t)b * E_DIM + e) * H_DIM + 8 * half;
        v16bf b2a, b2bf;
#pragma unroll
        for (int j = 0; j < 16; j += 4) {
            f32x4 t0 = *(const f32x4*)(w2row + j + (j & 8));
            f32x4 t1 = *(const f32x4*)(w2row + 32 + j + (j & 8));
            b2a[j] = tobf(t0[0]);  b2a[j + 1] = tobf(t0[1]);
            b2a[j + 2] = tobf(t0[2]); b2a[j + 3] = tobf(t0[3]);
            b2bf[j] = tobf(t1[0]); b2bf[j + 1] = tobf(t1[1]);
            b2bf[j + 2] = tobf(t1[2]); b2bf[j + 3] = tobf(t1[3]);
        }
        v8f c = {};
        c = __builtin_amdgcn_wmma_f32_16x16x32_bf16(false, a2a, false, b2a,
                                                    (short)0, c, false, false);
        c = __builtin_amdgcn_wmma_f32_16x16x32_bf16(false, a2b, false, b2bf,
                                                    (short)0, c, false, false);
        const float bias = b2[b * E_DIM + e];
        float* op = obase + 16 * g2;
#pragma unroll
        for (int r = 0; r < 8; ++r) {
            op[r * E_DIM] = c[r] + bias;   // constant 128B-stride offsets
        }
    }
}

// ---------------------------------------------------------------------------
extern "C" void kernel_launch(void* const* d_in, const int* in_sizes, int n_in,
                              void* d_out, int out_size, void* d_ws, size_t ws_size,
                              hipStream_t stream) {
    const float* x  = (const float*)d_in[0];
    const float* Qw = (const float*)d_in[1];
    const float* W1 = (const float*)d_in[2];
    const float* b1 = (const float*)d_in[3];
    const float* W2 = (const float*)d_in[4];
    const float* b2 = (const float*)d_in[5];
    float* out = (float*)d_out;

    // workspace layout: [ q bf16 (B*S*E) | sumsq f32 (B*E) ]
    __hip_bfloat16* qws = (__hip_bfloat16*)d_ws;
    float* sumsq = (float*)((char*)d_ws +
                            (size_t)B_DIM * S_LEN * E_DIM * sizeof(__hip_bfloat16));

    const int nss = B_DIM * E_DIM;
    nm_zero<<<(nss + 255) / 256, 256, 0, stream>>>(sumsq, nss);

    dim3 grid(S_LEN / TILE_S, B_DIM);
    dim3 block(32 * NW);
    nm_pass1<<<grid, block, 0, stream>>>(x, Qw, qws, sumsq);
    nm_pass2<<<grid, block, 0, stream>>>(qws, sumsq, W1, b1, W2, b2, out);
}
